// LeNet5Binary_73005854097691
// MI455X (gfx1250) — 8-bit matrix operand order repaired and checked by reading, compiled, not run on hardware
//
#include <hip/hip_runtime.h>

typedef __attribute__((ext_vector_type(8))) int v8i;

#define BATCH 16384

static constexpr size_t POOL1_OFF  = 0;
static constexpr size_t SGN1_OFF   = 77070336;
static constexpr size_t POOL2_OFF  = 96337920;
static constexpr size_t SGN2_OFF   = 122552320;
static constexpr size_t FC_OFF     = 129892352;
static constexpr size_t STATS_OFF  = 137756672;
static constexpr size_t ALPHA_OFF  = STATS_OFF + 2048;
static constexpr size_t W2BIN_OFF  = ALPHA_OFF + 1024;
static constexpr size_t W3BIN_OFF  = W2BIN_OFF + 9728;
static constexpr size_t W4B_OFF    = W3BIN_OFF + 192000;
static constexpr size_t W2FRAG_OFF = W4B_OFF + 40448;
static constexpr size_t W3FRAG_OFF = W2FRAG_OFF + 3072;

#define NSTATS 296

static constexpr float INV_N1 = 1.0f / ((float)BATCH * 28.f * 28.f);
static constexpr float INV_N2 = 1.0f / ((float)BATCH * 14.f * 14.f);
static constexpr float INV_N3 = 1.0f / ((float)BATCH * 5.f * 5.f);
static constexpr float INV_N4 = 1.0f / (float)BATCH;

__device__ __forceinline__ float signf(float v) {
    return (v > 0.f) ? 1.f : ((v < 0.f) ? -1.f : 0.f);
}
__device__ __forceinline__ signed char signb(float v) {
    return (v > 0.f) ? (signed char)1 : ((v < 0.f) ? (signed char)-1 : (signed char)0);
}

__global__ void k_zero(float* __restrict__ stats) {
    int t = blockIdx.x * blockDim.x + threadIdx.x;
    if (t < NSTATS) stats[t] = 0.f;
}

__global__ void k0a_binweights(const float* __restrict__ w2, const float* __restrict__ w3,
                               const float* __restrict__ w4,
                               float* __restrict__ w2bin, float* __restrict__ w3bin,
                               float* __restrict__ w4b, float* __restrict__ alphas) {
    int t = threadIdx.x;
    if (t < 16) {
        const float* w = w2 + t * 150;
        float mean[25];
        #pragma unroll
        for (int p = 0; p < 25; ++p) {
            float s = 0.f;
            #pragma unroll
            for (int ci = 0; ci < 6; ++ci) s += w[ci * 25 + p];
            mean[p] = s * (1.f / 6.f);
        }
        float alpha = 0.f;
        #pragma unroll
        for (int ci = 0; ci < 6; ++ci)
            #pragma unroll
            for (int p = 0; p < 25; ++p)
                alpha += fabsf(fminf(fmaxf(w[ci * 25 + p] - mean[p], -1.f), 1.f));
        alpha *= 0.25f;
        alphas[t] = alpha;
        #pragma unroll
        for (int ci = 0; ci < 6; ++ci)
            #pragma unroll
            for (int p = 0; p < 25; ++p) {
                float cw = fminf(fmaxf(w[ci * 25 + p] - mean[p], -1.f), 1.f);
                w2bin[t * 150 + ci * 25 + p] = signf(cw) * alpha;
            }
    }
    if (t < 120) {
        const float* w = w3 + t * 400;
        float mean[25];
        #pragma unroll
        for (int p = 0; p < 25; ++p) {
            float s = 0.f;
            #pragma unroll
            for (int ci = 0; ci < 16; ++ci) s += w[ci * 25 + p];
            mean[p] = s * (1.f / 16.f);
        }
        float alpha = 0.f;
        for (int ci = 0; ci < 16; ++ci)
            #pragma unroll
            for (int p = 0; p < 25; ++p)
                alpha += fabsf(fminf(fmaxf(w[ci * 25 + p] - mean[p], -1.f), 1.f));
        alpha *= 0.25f;
        alphas[16 + t] = alpha;
        for (int ci = 0; ci < 16; ++ci)
            #pragma unroll
            for (int p = 0; p < 25; ++p) {
                float cw = fminf(fmaxf(w[ci * 25 + p] - mean[p], -1.f), 1.f);
                w3bin[t * 400 + ci * 25 + p] = signf(cw) * alpha;
            }
    }
    if (t < 84) {
        const float* w = w4 + t * 120;
        float s = 0.f;
        for (int f = 0; f < 120; ++f) s += w[f];
        float m = s * (1.f / 120.f);
        float alpha = 0.f;
        for (int f = 0; f < 120; ++f)
            alpha += fabsf(fminf(fmaxf(w[f] - m, -1.f), 1.f));
        alpha *= 0.5f;
        for (int f = 0; f < 120; ++f) {
            float cw = fminf(fmaxf(w[f] - m, -1.f), 1.f);
            w4b[t * 120 + f] = signf(cw) * alpha;
        }
    }
}

__global__ void k0b_pack(const float* __restrict__ w2bin, const float* __restrict__ w3bin,
                         signed char* __restrict__ w2frag, signed char* __restrict__ w3frag) {
    int tid = blockIdx.x * blockDim.x + threadIdx.x;
    if (tid < 96) {
        int step = tid >> 5, lane = tid & 31;
        int n = lane & 15, grp = lane >> 4;
        #pragma unroll
        for (int bb = 0; bb < 32; ++bb) {
            int k = step * 64 + ((bb >> 4) << 5) + grp * 16 + (bb & 15);
            signed char v = (k < 150) ? signb(w2bin[n * 150 + k]) : (signed char)0;
            w2frag[(size_t)tid * 32 + bb] = v;
        }
    }
    if (tid < 1792) {
        int nt = tid / 224, r = tid % 224;
        int step = r >> 5, lane = r & 31;
        int co = nt * 16 + (lane & 15), grp = lane >> 4;
        #pragma unroll
        for (int bb = 0; bb < 32; ++bb) {
            int k = step * 64 + ((bb >> 3) << 4) + grp * 8 + (bb & 7);
            signed char v = (co < 120 && k < 400) ? signb(w3bin[co * 400 + k]) : (signed char)0;
            w3frag[(size_t)tid * 32 + bb] = v;
        }
    }
}

__global__ void k1_conv1_stats(const float* __restrict__ x, const float* __restrict__ w1,
                               const float* __restrict__ b1, float* __restrict__ stats) {
    __shared__ float sw[150];
    __shared__ float sb[6];
    __shared__ float sh[12];
    if (threadIdx.x < 150) sw[threadIdx.x] = w1[threadIdx.x];
    if (threadIdx.x < 6) sb[threadIdx.x] = b1[threadIdx.x];
    if (threadIdx.x < 12) sh[threadIdx.x] = 0.f;
    __syncthreads();

    int tid = blockIdx.x * blockDim.x + threadIdx.x;
    int b = tid / 784, r = tid % 784, y = r / 28, xo = r % 28;
    const float* xb = x + (size_t)b * 1024 + y * 32 + xo;
    float in[25];
    #pragma unroll
    for (int ky = 0; ky < 5; ++ky)
        #pragma unroll
        for (int kx = 0; kx < 5; ++kx) in[ky * 5 + kx] = xb[ky * 32 + kx];

    float s[6], q[6];
    #pragma unroll
    for (int c = 0; c < 6; ++c) {
        float a = sb[c];
        #pragma unroll
        for (int t = 0; t < 25; ++t) a += in[t] * sw[c * 25 + t];
        s[c] = a;
        q[c] = a * a;
    }
    #pragma unroll
    for (int c = 0; c < 6; ++c)
        for (int off = 16; off >= 1; off >>= 1) {
            s[c] += __shfl_xor(s[c], off);
            q[c] += __shfl_xor(q[c], off);
        }
    if ((threadIdx.x & 31) == 0) {
        #pragma unroll
        for (int c = 0; c < 6; ++c) {
            atomicAdd(&sh[c], s[c]);
            atomicAdd(&sh[6 + c], q[c]);
        }
    }
    __syncthreads();
    if (threadIdx.x < 12) atomicAdd(&stats[threadIdx.x], sh[threadIdx.x]);
}

__global__ void k2_pool1(const float* __restrict__ x, const float* __restrict__ w1,
                         const float* __restrict__ b1, float* __restrict__ stats,
                         float* __restrict__ pool1) {
    __shared__ float sw[150];
    __shared__ float sb[6];
    __shared__ float sh[12];
    if (threadIdx.x < 150) sw[threadIdx.x] = w1[threadIdx.x];
    if (threadIdx.x < 6) sb[threadIdx.x] = b1[threadIdx.x];
    if (threadIdx.x < 12) sh[threadIdx.x] = 0.f;
    __syncthreads();

    int tid = blockIdx.x * blockDim.x + threadIdx.x;
    int b = tid / 196, r = tid % 196, py = r / 14, px = r % 14;
    const float* xb = x + (size_t)b * 1024 + (2 * py) * 32 + 2 * px;
    float in[36];
    #pragma unroll
    for (int iy = 0; iy < 6; ++iy)
        #pragma unroll
        for (int ix = 0; ix < 6; ++ix) in[iy * 6 + ix] = xb[iy * 32 + ix];

    float s2[6], q2[6];
    #pragma unroll
    for (int c = 0; c < 6; ++c) {
        float m = stats[c] * INV_N1;
        float var = stats[6 + c] * INV_N1 - m * m;
        float rstd = rsqrtf(var + 1e-4f);
        float pm = -1e30f;
        #pragma unroll
        for (int dy = 0; dy < 2; ++dy)
            #pragma unroll
            for (int dx = 0; dx < 2; ++dx) {
                float a = sb[c];
                #pragma unroll
                for (int ky = 0; ky < 5; ++ky)
                    #pragma unroll
                    for (int kx = 0; kx < 5; ++kx)
                        a += in[(dy + ky) * 6 + (dx + kx)] * sw[c * 25 + ky * 5 + kx];
                float yv = fmaxf((a - m) * rstd, 0.f);
                pm = fmaxf(pm, yv);
            }
        pool1[(size_t)(b * 6 + c) * 196 + py * 14 + px] = pm;
        s2[c] = pm;
        q2[c] = pm * pm;
    }
    #pragma unroll
    for (int c = 0; c < 6; ++c)
        for (int off = 16; off >= 1; off >>= 1) {
            s2[c] += __shfl_xor(s2[c], off);
            q2[c] += __shfl_xor(q2[c], off);
        }
    if ((threadIdx.x & 31) == 0) {
        #pragma unroll
        for (int c = 0; c < 6; ++c) {
            atomicAdd(&sh[c], s2[c]);
            atomicAdd(&sh[6 + c], q2[c]);
        }
    }
    __syncthreads();
    if (threadIdx.x < 12) atomicAdd(&stats[12 + threadIdx.x], sh[threadIdx.x]);
}

__global__ void k3_sign1(const float* __restrict__ pool1, const float* __restrict__ stats,
                         const float* __restrict__ g2, const float* __restrict__ bt2,
                         signed char* __restrict__ sgn1) {
    __shared__ float sm[6], ssc[6], sbt[6];
    if (threadIdx.x < 6) {
        int c = threadIdx.x;
        float m = stats[12 + c] * INV_N2;
        float var = stats[18 + c] * INV_N2 - m * m;
        sm[c] = m;
        ssc[c] = rsqrtf(var + 1e-4f) * g2[c];
        sbt[c] = bt2[c];
    }
    __syncthreads();
    int idx = blockIdx.x * blockDim.x + threadIdx.x;
    int ci = (idx / 196) % 6;
    float t = pool1[idx];
    float xn = (t - sm[ci]) * ssc[ci] + sbt[ci];
    sgn1[idx] = signb(xn);
}

template <int GRP>
__device__ __forceinline__ void build_a_conv2(const signed char* __restrict__ basep,
                                              v8i afr[3]) {
    #pragma unroll
    for (int step = 0; step < 3; ++step) {
        #pragma unroll
        for (int v = 0; v < 8; ++v) {
            unsigned int d = 0;
            #pragma unroll
            for (int jj = 0; jj < 4; ++jj) {
                const int k = step * 64 + ((v >> 2) << 5) + GRP * 16 +
                              ((v & 3) << 2) + jj;
                if (k < 150) {
                    const int ci = k / 25, rk = k % 25, ky = rk / 5, kx = rk % 5;
                    d |= ((unsigned int)(unsigned char)basep[ci * 196 + ky * 14 + kx])
                         << (8 * jj);
                }
            }
            afr[step][v] = (int)d;
        }
    }
}

__global__ void k4_conv2_wmma(const signed char* __restrict__ sgn1,
                              const signed char* __restrict__ w2frag,
                              const float* __restrict__ alphas,
                              float* __restrict__ pool2, float* __restrict__ stats) {
    __shared__ float sal[16];
    if (threadIdx.x < 16) sal[threadIdx.x] = alphas[threadIdx.x];
    __syncthreads();

    int gid = blockIdx.x * blockDim.x + threadIdx.x;
    int mtile = gid >> 5;
    int lane = threadIdx.x & 31;
    int grp = lane >> 4, lm = lane & 15;
    int row = mtile * 16 + lm;
    int cell = row >> 2, mem = row & 3;
    int b = cell / 25, cr = cell % 25;
    int y = (cr / 5) * 2 + (mem >> 1);
    int xo = (cr % 5) * 2 + (mem & 1);
    const signed char* basep = sgn1 + (size_t)b * 1176 + y * 14 + xo;
    __builtin_prefetch(basep, 0, 1);

    v8i afr[3];
    if (grp == 0) build_a_conv2<0>(basep, afr);
    else          build_a_conv2<1>(basep, afr);
    v8i acc = {0, 0, 0, 0, 0, 0, 0, 0};
    #pragma unroll
    for (int step = 0; step < 3; ++step) {
        v8i bf = *(const v8i*)(w2frag + (size_t)(step * 32 + lane) * 32);
        acc = __builtin_amdgcn_wmma_i32_16x16x64_iu8(true, afr[step], true, bf, acc,
                                                     false, false);
    }
    float alpha = sal[lm];
    float ls = 0.f, lq = 0.f;
    #pragma unroll
    for (int c = 0; c < 2; ++c) {
        float pm = -1e30f;
        #pragma unroll
        for (int mI = 0; mI < 4; ++mI)
            pm = fmaxf(pm, fmaxf((float)acc[c * 4 + mI] * alpha, 0.f));
        int gcell = mtile * 4 + grp * 2 + c;
        int bb = gcell / 25, cc = gcell % 25;
        pool2[(size_t)(bb * 16 + lm) * 25 + cc] = pm;
        ls += pm;
        lq += pm * pm;
    }
    ls += __shfl_xor(ls, 16);
    lq += __shfl_xor(lq, 16);
    if (lane < 16) {
        atomicAdd(&stats[24 + lm], ls);
        atomicAdd(&stats[40 + lm], lq);
    }
}

__global__ void k5b_sign2(const float* __restrict__ pool2, const float* __restrict__ stats,
                          const float* __restrict__ g3, const float* __restrict__ bt3,
                          signed char* __restrict__ sgn2) {
    __shared__ float sm[16], ssc[16], sbt[16];
    if (threadIdx.x < 16) {
        int c = threadIdx.x;
        float m = stats[24 + c] * INV_N3;
        float var = stats[40 + c] * INV_N3 - m * m;
        sm[c] = m;
        ssc[c] = rsqrtf(var + 1e-4f) * g3[c];
        sbt[c] = bt3[c];
    }
    __syncthreads();
    int tid = blockIdx.x * blockDim.x + threadIdx.x;
    int b = tid / 448, p = tid % 448;
    signed char v = 0;
    if (p < 400) {
        int ci = p / 25;
        float t = pool2[(size_t)b * 400 + p];
        float xn = (t - sm[ci]) * ssc[ci] + sbt[ci];
        v = signb(xn);
    }
    sgn2[tid] = v;
}

__global__ void k6_conv3_wmma(const signed char* __restrict__ sgn2,
                              const signed char* __restrict__ w3frag,
                              const float* __restrict__ alphas,
                              float* __restrict__ stats, float* __restrict__ fc) {
    __shared__ float sal[120];
    if (threadIdx.x < 120) sal[threadIdx.x] = alphas[16 + threadIdx.x];
    __syncthreads();
    float* s4sum = stats + 56;

    int gid = blockIdx.x * blockDim.x + threadIdx.x;
    int mtile = gid >> 5;
    int lane = threadIdx.x & 31;
    int grp = lane >> 4, lm = lane & 15;
    int b = mtile * 16 + lm;
    const signed char* srow = sgn2 + (size_t)b * 448 + grp * 8;
    __builtin_prefetch(srow, 0, 1);

    v8i acc[8];
    #pragma unroll
    for (int nt = 0; nt < 8; ++nt) acc[nt] = (v8i){0, 0, 0, 0, 0, 0, 0, 0};

    for (int step = 0; step < 7; ++step) {
        const signed char* p = srow + step * 64;
        v8i a;
        a[0] = *(const int*)(p + 0);
        a[1] = *(const int*)(p + 4);
        a[2] = *(const int*)(p + 16);
        a[3] = *(const int*)(p + 20);
        a[4] = *(const int*)(p + 32);
        a[5] = *(const int*)(p + 36);
        a[6] = *(const int*)(p + 48);
        a[7] = *(const int*)(p + 52);
        #pragma unroll
        for (int nt = 0; nt < 8; ++nt) {
            v8i bf = *(const v8i*)(w3frag + (size_t)((nt * 7 + step) * 32 + lane) * 32);
            acc[nt] = __builtin_amdgcn_wmma_i32_16x16x64_iu8(true, a, true, bf, acc[nt],
                                                             false, false);
        }
    }
    #pragma unroll
    for (int nt = 0; nt < 8; ++nt) {
        int co = nt * 16 + lm;
        float alpha = (co < 120) ? sal[co] : 0.f;
        float ls = 0.f, lq = 0.f;
        #pragma unroll
        for (int rI = 0; rI < 8; ++rI) {
            int bb = mtile * 16 + grp * 8 + rI;
            float v = fmaxf((float)acc[nt][rI] * alpha, 0.f);
            if (co < 120) fc[(size_t)bb * 120 + co] = v;
            ls += v;
            lq += v * v;
        }
        ls += __shfl_xor(ls, 16);
        lq += __shfl_xor(lq, 16);
        if (lane < 16 && co < 120) {
            atomicAdd(&s4sum[co], ls);
            atomicAdd(&s4sum[120 + co], lq);
        }
    }
}

__global__ __launch_bounds__(128) void k7_fc(const float* __restrict__ fc,
                                             const float* __restrict__ stats,
                                             const float* __restrict__ g4,
                                             const float* __restrict__ bt4,
                                             const float* __restrict__ w4b,
                                             const float* __restrict__ w5,
                                             const float* __restrict__ b5,
                                             float* __restrict__ out) {
    __shared__ float sxn[128 * 120];
    const float* s4 = stats + 56;
    int b = blockIdx.x * blockDim.x + threadIdx.x;
    const float* xin = fc + (size_t)b * 120;
    float* myxn = sxn + threadIdx.x * 120;
    for (int f = 0; f < 120; ++f) {
        float m = s4[f] * INV_N4;
        float var = s4[120 + f] * INV_N4 - m * m;
        float rstd = rsqrtf(var + 1e-5f);
        myxn[f] = (xin[f] - m) * rstd * g4[f] + bt4[f];
    }
    float o[10];
    #pragma unroll
    for (int c = 0; c < 10; ++c) o[c] = b5[c];
    for (int j = 0; j < 84; ++j) {
        float h = 0.f;
        for (int f = 0; f < 120; ++f) h += myxn[f] * w4b[j * 120 + f];
        h = fmaxf(h, 0.f);
        #pragma unroll
        for (int c = 0; c < 10; ++c) o[c] += h * w5[c * 84 + j];
    }
    #pragma unroll
    for (int c = 0; c < 10; ++c) out[(size_t)b * 10 + c] = o[c];
}

extern "C" void kernel_launch(void* const* d_in, const int* in_sizes, int n_in,
                              void* d_out, int out_size, void* d_ws, size_t ws_size,
                              hipStream_t stream) {
    (void)in_sizes; (void)n_in; (void)out_size; (void)ws_size;
    const float* x   = (const float*)d_in[0];
    const float* w1  = (const float*)d_in[1];
    const float* b1  = (const float*)d_in[2];
    const float* g2  = (const float*)d_in[3];
    const float* bt2 = (const float*)d_in[4];
    const float* w2  = (const float*)d_in[5];
    const float* g3  = (const float*)d_in[6];
    const float* bt3 = (const float*)d_in[7];
    const float* w3  = (const float*)d_in[8];
    const float* g4  = (const float*)d_in[9];
    const float* bt4 = (const float*)d_in[10];
    const float* w4  = (const float*)d_in[11];
    const float* w5  = (const float*)d_in[12];
    const float* b5  = (const float*)d_in[13];
    float* out = (float*)d_out;

    char* ws = (char*)d_ws;
    float*        pool1  = (float*)(ws + POOL1_OFF);
    signed char*  sgn1   = (signed char*)(ws + SGN1_OFF);
    float*        pool2  = (float*)(ws + POOL2_OFF);
    signed char*  sgn2   = (signed char*)(ws + SGN2_OFF);
    float*        fcbuf  = (float*)(ws + FC_OFF);
    float*        stats  = (float*)(ws + STATS_OFF);
    float*        alphas = (float*)(ws + ALPHA_OFF);
    float*        w2bin  = (float*)(ws + W2BIN_OFF);
    float*        w3bin  = (float*)(ws + W3BIN_OFF);
    float*        w4b    = (float*)(ws + W4B_OFF);
    signed char*  w2frag = (signed char*)(ws + W2FRAG_OFF);
    signed char*  w3frag = (signed char*)(ws + W3FRAG_OFF);

    k_zero<<<2, 256, 0, stream>>>(stats);
    k0a_binweights<<<1, 128, 0, stream>>>(w2, w3, w4, w2bin, w3bin, w4b, alphas);
    k0b_pack<<<7, 256, 0, stream>>>(w2bin, w3bin, w2frag, w3frag);

    k1_conv1_stats<<<50176, 256, 0, stream>>>(x, w1, b1, stats);
    k2_pool1<<<12544, 256, 0, stream>>>(x, w1, b1, stats, pool1);
    k3_sign1<<<75264, 256, 0, stream>>>(pool1, stats, g2, bt2, sgn1);
    k4_conv2_wmma<<<25600, 128, 0, stream>>>(sgn1, w2frag, alphas, pool2, stats);
    k5b_sign2<<<28672, 256, 0, stream>>>(pool2, stats, g3, bt3, sgn2);
    k6_conv3_wmma<<<256, 128, 0, stream>>>(sgn2, w3frag, alphas, stats, fcbuf);
    k7_fc<<<128, 128, 0, stream>>>(fcbuf, stats, g4, bt4, w4b, w5, b5, out);
}
